// GNN_gat_50113678409874
// MI455X (gfx1250) — compile-verified, matrix operand order checked by reading, not run on hardware
//
#include <hip/hip_runtime.h>
#include <hip/hip_bf16.h>

#define HIDDEN 100
#define HEADS  8
#define LAYERS 3
#define ALPHA  0.2f
#define D      800
#define FEAT   32

typedef __attribute__((ext_vector_type(16))) __bf16 v16bf;
typedef __attribute__((ext_vector_type(8)))  __bf16 v8bf;
typedef __attribute__((ext_vector_type(4)))  __bf16 v4bf;
typedef __attribute__((ext_vector_type(8)))  float  v8f;
typedef __attribute__((ext_vector_type(4)))  float  v4f;

// gfx1250 async global->LDS path (guarded so compile never breaks)
#if defined(__gfx1250__) && __has_builtin(__builtin_amdgcn_global_load_async_to_lds_b128) && __has_builtin(__builtin_amdgcn_s_wait_asynccnt)
#define ASYNC_LDS 1
typedef int v4i __attribute__((__vector_size__(16)));
typedef __attribute__((address_space(1))) v4i* gas1_v4i;  // global int4*
typedef __attribute__((address_space(3))) v4i* las3_v4i;  // LDS int4*
#else
#define ASYNC_LDS 0
#endif

// ---------------------------------------------------------------- conversions

__global__ void f32_to_bf16_kernel(const float* __restrict__ src,
                                   __bf16* __restrict__ dstp, int count) {
    int i = blockIdx.x * blockDim.x + threadIdx.x;
    if (i < count) dstp[i] = (__bf16)src[i];
}

// generic f32 (R x C) -> bf16 transposed (C x R)
__global__ void transpose_to_bf16_kernel(const float* __restrict__ src,
                                         __bf16* __restrict__ dstp, int R, int C) {
    int i = blockIdx.x * blockDim.x + threadIdx.x;
    if (i >= R * C) return;
    int r = i / C, c = i - r * C;          // coalesced read over c
    dstp[(size_t)c * R + r] = (__bf16)src[i];
}

// kernels (L,H,D,HIDDEN) f32 -> per-layer N-major bf16: Bt[l][(h*100+c)*D + k]
__global__ void kernels_to_bf16_t_kernel(const float* __restrict__ ker,
                                         __bf16* __restrict__ Bt, int total) {
    int i = blockIdx.x * blockDim.x + threadIdx.x;
    if (i >= total) return;
    int c = i % HIDDEN;
    int k = (i / HIDDEN) % D;
    int h = (i / (HIDDEN * D)) % HEADS;
    int l = i / (HIDDEN * D * HEADS);
    Bt[(size_t)l * D * D + (size_t)(h * HIDDEN + c) * D + k] = (__bf16)ker[i];
}

// CSR row pointers from sorted src (lower_bound per node)
__global__ void rowptr_kernel(const int* __restrict__ src,
                              int* __restrict__ row_ptr, int n, int E) {
    int i = blockIdx.x * blockDim.x + threadIdx.x;
    if (i > n) return;
    if (i == n) { row_ptr[n] = E; return; }
    int lo = 0, hi = E;
    while (lo < hi) {
        int mid = (lo + hi) >> 1;
        if (src[mid] < i) lo = mid + 1; else hi = mid;
    }
    row_ptr[i] = lo;
}

// ---------------------------------------------------------------- WMMA GEMM
// C(MxN,f32) = A(MxK,bf16,row-major) * Bt(NxK,bf16: B stored column-major)
// Block: 256 threads = 8 waves. Tile 256(M) x 80(N), K-step 32, double-buffered
// LDS with async global->LDS prefetch of tile k+1 overlapped with WMMAs on tile k.

#define BLK_M 256
#define BLK_N 80
#define BLK_K 32
#define A_STR 40    // LDS row stride in bf16 (32 + 8 pad, keeps 16B alignment)
#define BT_STR 40

// stage one K-tile into the given LDS buffers (row-clamped: no exec branches)
__device__ __forceinline__
void stage_tile(const __bf16* __restrict__ A, const __bf16* __restrict__ Bt,
                __bf16* AsBuf, __bf16* BtsBuf,
                int m0, int n0, int k0, int M, int K, int tid) {
    // A tile: 256x32 bf16 = 1024 chunks of 8 -> 4 per thread
#pragma unroll
    for (int it = 0; it < 4; ++it) {
        int q = tid + it * 256;
        int rr = q >> 2, coloff = (q & 3) * 8;
        int gr = min(m0 + rr, M - 1);                 // clamp ragged tail
        const __bf16* gp = A + (size_t)gr * K + k0 + coloff;
        __bf16* lp = &AsBuf[rr * A_STR + coloff];
#if ASYNC_LDS
        __builtin_amdgcn_global_load_async_to_lds_b128(
            (gas1_v4i)(void*)gp, (las3_v4i)(void*)lp, 0, 0);
#else
        *(v8bf*)lp = *(const v8bf*)gp;
#endif
    }
    // B tile: 80 (cols of B) x 32 K = 320 chunks of 8
    {
        int rr = tid >> 2, coloff = (tid & 3) * 8;
        const __bf16* gp = Bt + (size_t)(n0 + rr) * K + k0 + coloff;
        __bf16* lp = &BtsBuf[rr * BT_STR + coloff];
#if ASYNC_LDS
        __builtin_amdgcn_global_load_async_to_lds_b128(
            (gas1_v4i)(void*)gp, (las3_v4i)(void*)lp, 0, 0);
#else
        *(v8bf*)lp = *(const v8bf*)gp;
#endif
    }
    if (tid < 64) {
        int q = 256 + tid;
        int rr = q >> 2, coloff = (q & 3) * 8;
        const __bf16* gp = Bt + (size_t)(n0 + rr) * K + k0 + coloff;
        __bf16* lp = &BtsBuf[rr * BT_STR + coloff];
#if ASYNC_LDS
        __builtin_amdgcn_global_load_async_to_lds_b128(
            (gas1_v4i)(void*)gp, (las3_v4i)(void*)lp, 0, 0);
#else
        *(v8bf*)lp = *(const v8bf*)gp;
#endif
    }
}

__global__ __launch_bounds__(256)
void gemm_bf16_kernel(const __bf16* __restrict__ A, const __bf16* __restrict__ Bt,
                      float* __restrict__ C, __bf16* __restrict__ Cb,
                      const float* __restrict__ bias, int M, int N, int K, int relu) {
    __shared__ __align__(16) __bf16 As [2][BLK_M * A_STR];
    __shared__ __align__(16) __bf16 Bts[2][BLK_N * BT_STR];

    const int tid  = threadIdx.x;
    const int wave = tid >> 5;
    const int lane = tid & 31;
    const int hlf  = lane >> 4;
    const int r    = lane & 15;

    const int m0 = blockIdx.x * BLK_M;
    const int n0 = blockIdx.y * BLK_N;

    v8f acc[2][5];
#pragma unroll
    for (int m = 0; m < 2; ++m)
#pragma unroll
        for (int sb = 0; sb < 5; ++sb)
            acc[m][sb] = (v8f){0.f,0.f,0.f,0.f,0.f,0.f,0.f,0.f};

    // prologue: stage tile 0 into buffer 0
    stage_tile(A, Bt, As[0], Bts[0], m0, n0, 0, M, K, tid);
#if ASYNC_LDS
    __builtin_amdgcn_s_wait_asynccnt(0);
#endif
    __syncthreads();

    const int nsteps = K / BLK_K;
    for (int step = 0; step < nsteps; ++step) {
        const int cur = step & 1;
        // prefetch next tile into the alternate buffer while computing this one
        if (step + 1 < nsteps)
            stage_tile(A, Bt, As[cur ^ 1], Bts[cur ^ 1],
                       m0, n0, (step + 1) * BLK_K, M, K, tid);

        const __bf16* AsC  = As[cur];
        const __bf16* BtsC = Bts[cur];

        // A fragments: per-lane two contiguous 16B runs -> 2x ds_load_b128 each
        v16bf af[2];
#pragma unroll
        for (int m = 0; m < 2; ++m) {
            int rowI = wave * 32 + m * 16 + r;
            v8bf lo = *(const v8bf*)(&AsC[rowI * A_STR + hlf * 8]);
            v8bf hi = *(const v8bf*)(&AsC[rowI * A_STR + 16 + hlf * 8]);
            af[m] = __builtin_shufflevector(lo, hi, 0,1,2,3,4,5,6,7,8,9,10,11,12,13,14,15);
        }
#pragma unroll
        for (int sb = 0; sb < 5; ++sb) {
            int col = sb * 16 + r;
            v8bf lo = *(const v8bf*)(&BtsC[col * BT_STR + hlf * 8]);
            v8bf hi = *(const v8bf*)(&BtsC[col * BT_STR + 16 + hlf * 8]);
            v16bf bfr = __builtin_shufflevector(lo, hi, 0,1,2,3,4,5,6,7,8,9,10,11,12,13,14,15);
            acc[0][sb] = __builtin_amdgcn_wmma_f32_16x16x32_bf16(
                false, af[0], false, bfr, (short)0, acc[0][sb], false, false);
            acc[1][sb] = __builtin_amdgcn_wmma_f32_16x16x32_bf16(
                false, af[1], false, bfr, (short)0, acc[1][sb], false, false);
        }
#if ASYNC_LDS
        __builtin_amdgcn_s_wait_asynccnt(0);   // next-tile transfers complete
#endif
        __syncthreads();                       // all waves done reading cur
    }

    // epilogue: C layout — lane gives col r, VGPR j gives row hlf*8+j
#pragma unroll
    for (int m = 0; m < 2; ++m)
#pragma unroll
        for (int sb = 0; sb < 5; ++sb)
#pragma unroll
            for (int j = 0; j < 8; ++j) {
                int gr = m0 + wave * 32 + m * 16 + hlf * 8 + j;
                int gc = n0 + sb * 16 + r;
                if (gr < M) {
                    float v = acc[m][sb][j];
                    if (bias) v += bias[gc];
                    if (relu) v = fmaxf(v, 0.0f);
                    C[(size_t)gr * N + gc] = v;
                    if (Cb) Cb[(size_t)gr * N + gc] = (__bf16)v;
                }
            }
}

// ------------------------------------------------- attention s/t projections
__global__ __launch_bounds__(256)
void st_kernel(const float* __restrict__ H, const float* __restrict__ att_l,
               float* __restrict__ s, float* __restrict__ t, int n) {
    int node = blockIdx.x;
    int head = threadIdx.x >> 5;
    int lane = threadIdx.x & 31;
    if (node >= n) return;
    const float* hrow = H + (size_t)node * D + head * HIDDEN;
    const float* a1 = att_l + head * (2 * HIDDEN);
    const float* a2 = a1 + HIDDEN;
    float ps = 0.f, pt = 0.f;
    if (lane < HIDDEN / 4) {                       // lanes 0..24, float4 each
        v4f h = *(const v4f*)(hrow + lane * 4);
        v4f x1 = *(const v4f*)(a1 + lane * 4);
        v4f x2 = *(const v4f*)(a2 + lane * 4);
#pragma unroll
        for (int k = 0; k < 4; ++k) { ps += h[k] * x1[k]; pt += h[k] * x2[k]; }
    }
#pragma unroll
    for (int off = 16; off > 0; off >>= 1) {
        ps += __shfl_down(ps, off, 32);
        pt += __shfl_down(pt, off, 32);
    }
    if (lane == 0) {
        s[node * HEADS + head] = ps;
        t[node * HEADS + head] = pt;
    }
}

// ------------------------------------------------- segment softmax + scatter
__global__ __launch_bounds__(256)
void agg_kernel(const float* __restrict__ H, const int* __restrict__ dst,
                const int* __restrict__ row_ptr,
                const float* __restrict__ s, const float* __restrict__ t,
                float* __restrict__ x, __bf16* __restrict__ xb, int n) {
    int node = blockIdx.x;
    int head = threadIdx.x >> 5;
    int lane = threadIdx.x & 31;
    if (node >= n) return;
    int start = row_ptr[node];
    int end   = row_ptr[node + 1];
    float si = s[node * HEADS + head];

    // pass 1: softmax denominator
    float wsum = 0.f;
    for (int base = start; base < end; base += 32) {
        int e = base + lane;
        if (e < end) {
            int dj = dst[e];
            float v = si + t[dj * HEADS + head];
            v = (v > 0.f) ? v : ALPHA * v;           // leaky_relu
            v = fminf(fmaxf(v, -2.f), 2.f);          // clip
            wsum += __expf(v);
        }
    }
#pragma unroll
    for (int off = 16; off > 0; off >>= 1) wsum += __shfl_down(wsum, off, 32);
    float denom = __shfl(wsum, 0, 32);
    float inv = (end > start) ? 1.f / denom : 0.f;

    // pass 2: weighted gather of neighbor features
    v4f acc = (v4f){0.f, 0.f, 0.f, 0.f};
    for (int base = start; base < end; base += 32) {
        int cnt = min(32, end - base);
        int e = base + lane;
        float wl = 0.f; int dl = 0;
        if (e < end) {
            dl = dst[e];
            __builtin_prefetch(H + (size_t)dl * D + head * HIDDEN, 0, 0);
            float v = si + t[dl * HEADS + head];
            v = (v > 0.f) ? v : ALPHA * v;
            v = fminf(fmaxf(v, -2.f), 2.f);
            wl = __expf(v);
        }
        for (int j = 0; j < cnt; ++j) {
            float w = __shfl(wl, j, 32);
            int dj  = __shfl(dl, j, 32);
            if (lane < HIDDEN / 4) {
                v4f hv = *(const v4f*)(H + (size_t)dj * D + head * HIDDEN + lane * 4);
#pragma unroll
                for (int k = 0; k < 4; ++k) acc[k] += w * hv[k];
            }
        }
    }

    if (lane < HIDDEN / 4) {
        size_t rbase = (size_t)node * D + head * HIDDEN + lane * 4;
        v4f oldv = *(const v4f*)(x + rbase);
        v4f nv; v4bf bv;
#pragma unroll
        for (int k = 0; k < 4; ++k) {
            nv[k] = fmaxf(acc[k] * inv, 0.f) + oldv[k];   // relu + residual
            bv[k] = (__bf16)nv[k];
        }
        *(v4f*) (x  + rbase) = nv;
        *(v4bf*)(xb + rbase) = bv;
    }
}

// ---------------------------------------------------------------- final head
__global__ __launch_bounds__(256)
void logits_kernel(const float* __restrict__ x, const float* __restrict__ Wl,
                   const float* __restrict__ bl, float* __restrict__ logits, int n) {
    int node = blockIdx.x * 8 + (threadIdx.x >> 5);
    int lane = threadIdx.x & 31;
    if (node >= n) return;
    const float* row = x + (size_t)node * D;
    float p = 0.f;
    for (int cc = lane; cc < D / 4; cc += 32) {       // 200 float4 chunks
        v4f rv = *(const v4f*)(row + cc * 4);
        v4f wv = *(const v4f*)(Wl + cc * 4);
#pragma unroll
        for (int k = 0; k < 4; ++k) p += rv[k] * wv[k];
    }
#pragma unroll
    for (int off = 16; off > 0; off >>= 1) p += __shfl_down(p, off, 32);
    if (lane == 0) logits[node] = p + bl[0];
}

__global__ __launch_bounds__(512)
void reduce_kernel(const float* __restrict__ logits, const float* __restrict__ Wc,
                   const float* __restrict__ bc, float* __restrict__ out, int n) {
    __shared__ float sdata[512];
    float p = 0.f;
    for (int i = threadIdx.x; i < n; i += 512) p += logits[i] * Wc[i];
    sdata[threadIdx.x] = p;
    __syncthreads();
    for (int sft = 256; sft > 0; sft >>= 1) {
        if ((int)threadIdx.x < sft) sdata[threadIdx.x] += sdata[threadIdx.x + sft];
        __syncthreads();
    }
    if (threadIdx.x == 0) out[0] = sdata[0] + bc[0];
}

// ---------------------------------------------------------------- launch

extern "C" void kernel_launch(void* const* d_in, const int* in_sizes, int n_in,
                              void* d_out, int out_size, void* d_ws, size_t ws_size,
                              hipStream_t stream) {
    const float* node_states = (const float*)d_in[0];
    const int*   src         = (const int*)  d_in[1];
    const int*   dst         = (const int*)  d_in[2];
    const float* W_pre       = (const float*)d_in[3];
    const float* b_pre       = (const float*)d_in[4];
    const float* kernels     = (const float*)d_in[5];
    const float* att         = (const float*)d_in[6];
    const float* Wl          = (const float*)d_in[7];
    const float* bl          = (const float*)d_in[8];
    const float* Wc          = (const float*)d_in[9];
    const float* bc          = (const float*)d_in[10];

    const int n = in_sizes[0] / FEAT;
    const int E = in_sizes[1];

    // bump allocator on workspace (256B aligned)
    char* ws = (char*)d_ws;
    size_t off = 0;
    auto alloc = [&](size_t bytes) -> void* {
        void* p = ws + off;
        off += (bytes + 255) & ~(size_t)255;
        return p;
    };
    float*  x_f32 = (float*) alloc((size_t)n * D * sizeof(float));
    __bf16* x_bf  = (__bf16*)alloc((size_t)n * D * sizeof(__bf16));
    float*  Hall  = (float*) alloc((size_t)n * D * sizeof(float));
    __bf16* Bt    = (__bf16*)alloc((size_t)LAYERS * D * D * sizeof(__bf16)); // N-major
    __bf16* Apre  = (__bf16*)alloc((size_t)n * FEAT * sizeof(__bf16));
    __bf16* Wpret = (__bf16*)alloc((size_t)D * FEAT * sizeof(__bf16));       // 800x32
    float*  sArr  = (float*) alloc((size_t)n * HEADS * sizeof(float));
    float*  tArr  = (float*) alloc((size_t)n * HEADS * sizeof(float));
    int*    rowp  = (int*)   alloc((size_t)(n + 1) * sizeof(int));
    float*  lg    = (float*) alloc((size_t)n * sizeof(float));

    // input conversions (weights pre-transposed to N-major for vector LDS path)
    int cnt = n * FEAT;
    f32_to_bf16_kernel<<<(cnt + 255) / 256, 256, 0, stream>>>(node_states, Apre, cnt);
    cnt = FEAT * D;
    transpose_to_bf16_kernel<<<(cnt + 255) / 256, 256, 0, stream>>>(W_pre, Wpret, FEAT, D);
    int tot = LAYERS * HEADS * D * HIDDEN;
    kernels_to_bf16_t_kernel<<<(tot + 255) / 256, 256, 0, stream>>>(kernels, Bt, tot);
    rowptr_kernel<<<(n + 256) / 256, 256, 0, stream>>>(src, rowp, n, E);

    dim3 ggrid((n + BLK_M - 1) / BLK_M, D / BLK_N);

    // x = relu(NS @ W_pre + b_pre)   (writes f32 + bf16 copy)
    gemm_bf16_kernel<<<ggrid, 256, 0, stream>>>(Apre, Wpret, x_f32, x_bf,
                                                b_pre, n, D, FEAT, 1);

    for (int l = 0; l < LAYERS; ++l) {
        gemm_bf16_kernel<<<ggrid, 256, 0, stream>>>(x_bf, Bt + (size_t)l * D * D,
                                                    Hall, nullptr, nullptr, n, D, D, 0);
        st_kernel<<<n, 256, 0, stream>>>(Hall, att + (size_t)l * HEADS * 2 * HIDDEN,
                                         sArr, tArr, n);
        agg_kernel<<<n, 256, 0, stream>>>(Hall, dst, rowp, sArr, tArr, x_f32, x_bf, n);
    }

    logits_kernel<<<(n + 7) / 8, 256, 0, stream>>>(x_f32, Wl, bl, lg, n);
    reduce_kernel<<<1, 512, 0, stream>>>(lg, Wc, bc, (float*)d_out, n);
}
